// TextGenerationModel_22960895165169
// MI455X (gfx1250) — compile-verified, matrix operand order checked by reading, not run on hardware
//
#include <hip/hip_runtime.h>
#include <hip/hip_bf16.h>

#define VOCAB   128
#define EMBED   256
#define HIDDEN  1024
#define SEQ     512
#define BATCH   128
#define KDIM    (EMBED + HIDDEN)      // 1280
#define KT      40                    // K tiles (32 wide) in A / Wcat
#define NCAT    (4 * HIDDEN)          // 4096
#define TILE_ELEMS 512                // 16x32 bf16 tile
#define A_STRIDE (BATCH * KDIM)       // 163840 bf16 per step

#define TDM_CHUNK  8                       // K-tiles staged per TDM op
#define TDM_ELEMS  (TDM_CHUNK * 512)       // 4096 bf16 = 8KB per gate chunk
#define NCHUNK     (KT / TDM_CHUNK)        // 5

typedef __bf16  bf16x16 __attribute__((ext_vector_type(16)));
typedef float   f32x8   __attribute__((ext_vector_type(8)));
typedef unsigned int u32x4 __attribute__((ext_vector_type(4)));
typedef int          i32x4 __attribute__((ext_vector_type(4)));
typedef int          i32x8 __attribute__((ext_vector_type(8)));

__device__ __forceinline__ unsigned short f2bf(float f) {
    union { float f; unsigned u; } v; v.f = f;
    unsigned r = v.u + 0x7FFFu + ((v.u >> 16) & 1u);   // round-to-nearest-even
    return (unsigned short)(r >> 16);
}

__device__ __forceinline__ f32x8 bfwmma(bf16x16 a, bf16x16 b, f32x8 c) {
    return __builtin_amdgcn_wmma_f32_16x16x32_bf16(
        false, a, false, b, (short)0, c, false, false);
}

__device__ __forceinline__ float sigmoidf(float x) { return 1.0f / (1.0f + __expf(-x)); }

// ---- Tensor Data Mover: 1-D contiguous copy of one 8KB weight chunk to LDS ----
// D# per ISA 8.3/8.4: group0 = {count=1 | lds_addr | global_addr | type=2},
// group1 = {data_size=2B, tensor_dim0=tile_dim0=4096, dim1=1, stride0=4096}.
__device__ __forceinline__ void tdm_issue_chunk(const unsigned short* Wsrc, int ntg, int chunk,
                                                unsigned ldsByteOff) {
    unsigned long long ga = (unsigned long long)(const void*)Wsrc
        + ((unsigned long long)ntg * KT + (unsigned long long)chunk * TDM_CHUNK) * 1024ull;
    u32x4 g0;
    g0.x = 1u;                                                  // count=1 (valid)
    g0.y = ldsByteOff;                                          // lds_addr (bytes)
    g0.z = (unsigned)ga;                                        // global_addr[31:0]
    g0.w = (unsigned)((ga >> 32) & 0x01FFFFFFull) | (2u << 30); // addr[56:32] | type=2
    i32x8 g1;
    g1[0] = 0x00010000;                 // workgroup_mask=0, data_size=1 (2 bytes)
    g1[1] = (int)((unsigned)TDM_ELEMS << 16);   // tensor_dim0[15:0] in [31:16]
    g1[2] = 0x00010000;                 // tensor_dim0[31:16]=0 | tensor_dim1=1
    g1[3] = (int)((unsigned)TDM_ELEMS << 16);   // tile_dim0 in [31:16]
    g1[4] = 0;                          // tile_dim1=0 (unused), tile_dim2=0
    g1[5] = (int)TDM_ELEMS;             // tensor_dim0_stride[31:0]
    g1[6] = 0;
    g1[7] = 0;
    i32x4 z4 = (i32x4)0;
#if defined(__clang_major__) && (__clang_major__ >= 23)
    i32x8 z8 = (i32x8)0;
    __builtin_amdgcn_tensor_load_to_lds(g0, g1, z4, z4, z8, 0);
#else
    __builtin_amdgcn_tensor_load_to_lds(g0, g1, z4, z4, 0);
#endif
}

// --- packed tile addressing (ISA 7.12.2, 16-bit A 16x32 layout) ---
// lane l (0..31), elem j (0..15):  mLocal = l%16, half = l/16
//   j<8  -> kk = half*8 + j ;  j>=8 -> kk = 16 + half*8 + (j-8)

// ---------------------------------------------------------------- one-time packs
__global__ void k_embBf(const float* __restrict__ emb, unsigned short* __restrict__ embBf) {
    int i = blockIdx.x * blockDim.x + threadIdx.x;
    if (i < VOCAB * EMBED) embBf[i] = f2bf(emb[i]);
}

__global__ void k_bcat(const float* __restrict__ bi, const float* __restrict__ bf,
                       const float* __restrict__ bg, const float* __restrict__ bo,
                       float* __restrict__ bcat) {
    int i = blockIdx.x * blockDim.x + threadIdx.x;
    if (i >= NCAT) return;
    int g = i >> 10, col = i & 1023;
    const float* src = (g == 0) ? bi : (g == 1) ? bf : (g == 2) ? bg : bo;
    bcat[i] = src[col];
}

__global__ void k_packWcat(const float* __restrict__ Wix, const float* __restrict__ Wih,
                           const float* __restrict__ Wfx, const float* __restrict__ Wfh,
                           const float* __restrict__ Wgx, const float* __restrict__ Wgh,
                           const float* __restrict__ Wox, const float* __restrict__ Woh,
                           unsigned short* __restrict__ Wcat) {
    int p = blockIdx.x * blockDim.x + threadIdx.x;
    if (p >= KDIM * NCAT) return;
    int tIdx   = p >> 9;
    int within = p & 511;
    int l = within >> 4, j = within & 15;
    int nt = tIdx / KT, kt = tIdx % KT;
    int n    = nt * 16 + (l & 15);
    int half = l >> 4;
    int kk   = (j < 8) ? (half * 8 + j) : (16 + half * 8 + (j - 8));
    int k    = kt * 32 + kk;
    int g = n >> 10, col = n & 1023;
    const float* Wx = (g == 0) ? Wix : (g == 1) ? Wfx : (g == 2) ? Wgx : Wox;
    const float* Wh = (g == 0) ? Wih : (g == 1) ? Wfh : (g == 2) ? Wgh : Woh;
    float v = (k < EMBED) ? Wx[(size_t)k * HIDDEN + col]
                          : Wh[(size_t)(k - EMBED) * HIDDEN + col];
    Wcat[p] = f2bf(v);
}

__global__ void k_packWcls(const float* __restrict__ Wc, unsigned short* __restrict__ Wcls) {
    int p = blockIdx.x * blockDim.x + threadIdx.x;
    if (p >= HIDDEN * VOCAB) return;
    int tIdx = p >> 9, within = p & 511;
    int l = within >> 4, j = within & 15;
    int nt = tIdx >> 5, kt = tIdx & 31;
    int v    = nt * 16 + (l & 15);
    int half = l >> 4;
    int kk   = (j < 8) ? (half * 8 + j) : (16 + half * 8 + (j - 8));
    int h    = kt * 32 + kk;
    Wcls[p] = f2bf(Wc[(size_t)v * HIDDEN + h]);
}

__global__ void k_init(float* __restrict__ cbuf, unsigned short* __restrict__ Apk0) {
    int i = blockIdx.x * blockDim.x + threadIdx.x;
    if (i >= BATCH * HIDDEN) return;
    cbuf[i] = 0.0f;
    int mt = i >> 14;
    int rem = i & 16383;
    int kt = 8 + (rem >> 9);
    int within = rem & 511;
    Apk0[(size_t)(mt * KT + kt) * TILE_ELEMS + within] = 0;
}

__global__ void k_packAemb(const int* __restrict__ x, const unsigned short* __restrict__ embBf,
                           unsigned short* __restrict__ Apk) {
    size_t p = (size_t)blockIdx.x * blockDim.x + threadIdx.x;
    if (p >= (size_t)SEQ * 64 * 512) return;
    int t      = (int)(p >> 15);
    int rem    = (int)(p & 32767);
    int tl     = rem >> 9;
    int within = rem & 511;
    int mt = tl >> 3, kt = tl & 7;
    int l = within >> 4, j = within & 15;
    int m    = mt * 16 + (l & 15);
    int half = l >> 4;
    int kk   = (j < 8) ? (half * 8 + j) : (16 + half * 8 + (j - 8));
    int k    = kt * 32 + kk;
    int tok  = x[t * BATCH + m];
    Apk[(size_t)t * A_STRIDE + (size_t)(mt * KT + kt) * TILE_ELEMS + within] =
        embBf[tok * EMBED + k];
}

// ---------------------------------------------------------------- recurrence step
// One block (8 waves = 8 M-tiles) per hidden tile ht. Waves 0..3 TDM-stage the
// 4 gate weight streams into double-buffered LDS; all waves run the K=1280 GEMM
// with register-pipelined A loads, then the fused LSTM cell. h_t is written
// directly into the packed A layout of step t+1.
__global__ __launch_bounds__(256) void k_step(
        const unsigned short* __restrict__ ApkT,     // A_packed[t]
        unsigned short* __restrict__ ApkNext,        // A_packed[t+1]
        const unsigned short* __restrict__ Wcat,
        const float* __restrict__ bcat,
        float* __restrict__ cbuf) {
    __shared__ unsigned short sB[2][4][TDM_ELEMS];   // 64 KB of the 320 KB WGP LDS
    const int lane = threadIdx.x & 31;
    const int w    = threadIdx.x >> 5;     // 0..7  == mt
    const int mt   = w;
    const int ht   = blockIdx.x;           // 0..63

    const unsigned ldsBase = (unsigned)(unsigned long long)&sB[0][0][0];
    const int ntg = w * 64 + ht;           // gate-w weight stream (waves 0..3 only)

    const bf16x16* A = (const bf16x16*)ApkT + (size_t)mt * KT * 32 + lane;

    f32x8 ai = {}, af = {}, ag = {}, ao = {};
    bf16x16 a_cur = A[0];

    if (w < 4) {
        tdm_issue_chunk(Wcat, ntg, 0, ldsBase + (0 * 4 + w) * (TDM_ELEMS * 2));
        __builtin_amdgcn_s_wait_tensorcnt(0);
    }
    __syncthreads();

    int ktG = 0;
    for (int c = 0; c < NCHUNK; ++c) {
        if (w < 4 && (c + 1) < NCHUNK)
            tdm_issue_chunk(Wcat, ntg, c + 1,
                            ldsBase + ((((c + 1) & 1) * 4 + w) * (TDM_ELEMS * 2)));
        const unsigned short* bb = &sB[c & 1][0][0];
        #pragma unroll
        for (int kl = 0; kl < TDM_CHUNK; ++kl) {
            bf16x16 a_nxt = A[(ktG + 1) * 32];            // pipelined (pad-covered)
            const unsigned short* bk = bb + kl * 512 + lane * 16;
            bf16x16 b0 = *(const bf16x16*)(bk + 0 * TDM_ELEMS);
            bf16x16 b1 = *(const bf16x16*)(bk + 1 * TDM_ELEMS);
            bf16x16 b2 = *(const bf16x16*)(bk + 2 * TDM_ELEMS);
            bf16x16 b3 = *(const bf16x16*)(bk + 3 * TDM_ELEMS);
            ai = bfwmma(a_cur, b0, ai);
            af = bfwmma(a_cur, b1, af);
            ag = bfwmma(a_cur, b2, ag);
            ao = bfwmma(a_cur, b3, ao);
            a_cur = a_nxt;
            ++ktG;
        }
        if ((c + 1) < NCHUNK) {
            if (w < 4) __builtin_amdgcn_s_wait_tensorcnt(0);
            __syncthreads();
        }
    }

    // C/D layout: lanes 0-15 -> N=lane, M=r ; lanes 16-31 -> N=lane-16, M=8+r
    const int nl   = lane & 15;
    const int half = lane >> 4;
    const int hcol = ht * 16 + nl;
    const float bi  = bcat[hcol];
    const float bff = bcat[HIDDEN + hcol];
    const float bg  = bcat[2 * HIDDEN + hcol];
    const float bo  = bcat[3 * HIDDEN + hcol];

    // packed-A destination of h_t inside A_packed[t+1]
    const int ktn = 8 + (ht >> 1);
    const int kk  = ((ht & 1) << 4) + nl;
    int hlf, jh;
    if (kk < 8)       { hlf = 0; jh = kk; }
    else if (kk < 16) { hlf = 1; jh = kk - 8; }
    else if (kk < 24) { hlf = 0; jh = kk - 8; }
    else              { hlf = 1; jh = kk - 16; }
    unsigned short* hdst = ApkNext + (size_t)(mt * KT + ktn) * TILE_ELEMS + jh;

    #pragma unroll
    for (int r = 0; r < 8; ++r) {
        const int mloc = (half << 3) + r;
        const int m    = mt * 16 + mloc;
        const size_t cidx = (size_t)m * HIDDEN + hcol;
        float iv = sigmoidf(ai[r] + bi);
        float fv = sigmoidf(af[r] + bff);
        float gv = tanhf(ag[r] + bg);
        float ov = sigmoidf(ao[r] + bo);
        float cn = fv * cbuf[cidx] + iv * gv;
        cbuf[cidx] = cn;
        hdst[(size_t)(hlf * 16 + mloc) * 16] = f2bf(ov * tanhf(cn));
    }
}

// ---------------------------------------------------------------- classifier
__global__ __launch_bounds__(128) void k_cls(
        const unsigned short* __restrict__ Apk,
        const unsigned short* __restrict__ Wcls,
        const float* __restrict__ bcls,
        float* __restrict__ out) {
    const int lane = threadIdx.x & 31;
    const int wv   = blockIdx.x * 4 + (threadIdx.x >> 5);  // 0..32767
    const int t    = wv >> 6;
    const int mt   = (wv >> 3) & 7;
    const int nt   = wv & 7;

    const bf16x16* A = (const bf16x16*)(Apk + (size_t)(t + 1) * A_STRIDE)
                       + (size_t)(mt * KT + 8) * 32 + lane;
    const bf16x16* B = (const bf16x16*)Wcls + (size_t)(nt * 32) * 32 + lane;

    f32x8 acc = {};
    bf16x16 a = A[0], b = B[0];
    for (int kt = 0; kt < 32; ++kt) {
        bf16x16 an = A[(kt + 1) * 32];        // pipelined (pad-covered)
        bf16x16 bn = B[(kt + 1) * 32];
        acc = bfwmma(a, b, acc);
        a = an; b = bn;
    }

    const int nl = lane & 15, half = lane >> 4;
    const int v  = nt * 16 + nl;
    const float bias = bcls[v];
    const int mb = mt * 16 + (half << 3);
    #pragma unroll
    for (int r = 0; r < 8; ++r) {
        out[((size_t)t * BATCH + mb + r) * VOCAB + v] = acc[r] + bias;
    }
}

// ----------------------------------------------------------------------------
extern "C" void kernel_launch(void* const* d_in, const int* in_sizes, int n_in,
                              void* d_out, int out_size, void* d_ws, size_t ws_size,
                              hipStream_t stream) {
    const int*   x    = (const int*)  d_in[0];
    const float* emb  = (const float*)d_in[1];
    const float* W_ix = (const float*)d_in[2];
    const float* W_ih = (const float*)d_in[3];
    const float* b_i  = (const float*)d_in[4];
    const float* W_fx = (const float*)d_in[5];
    const float* W_fh = (const float*)d_in[6];
    const float* b_f  = (const float*)d_in[7];
    const float* W_gx = (const float*)d_in[8];
    const float* W_gh = (const float*)d_in[9];
    const float* b_g  = (const float*)d_in[10];
    const float* W_ox = (const float*)d_in[11];
    const float* W_oh = (const float*)d_in[12];
    const float* b_o  = (const float*)d_in[13];
    const float* W_c  = (const float*)d_in[14];
    const float* b_c  = (const float*)d_in[15];

    char* ws = (char*)d_ws;
    size_t off = 0;
    auto alloc = [&](size_t bytes) -> void* {
        void* p = ws + off;
        off = (off + bytes + 255) & ~(size_t)255;
        return p;
    };
    unsigned short* embBf = (unsigned short*)alloc((size_t)VOCAB * EMBED * 2);
    unsigned short* Wcat  = (unsigned short*)alloc((size_t)KDIM * NCAT * 2 + 2048);
    unsigned short* Wcls  = (unsigned short*)alloc((size_t)HIDDEN * VOCAB * 2 + 2048);
    float*          bcat  = (float*)         alloc((size_t)NCAT * 4);
    float*          cbuf  = (float*)         alloc((size_t)BATCH * HIDDEN * 4);
    unsigned short* Apk   = (unsigned short*)alloc((size_t)(SEQ + 1) * A_STRIDE * 2 + 2048);

    k_embBf   <<<(VOCAB * EMBED + 255) / 256, 256, 0, stream>>>(emb, embBf);
    k_bcat    <<<(NCAT + 255) / 256, 256, 0, stream>>>(b_i, b_f, b_g, b_o, bcat);
    k_packWcat<<<(KDIM * NCAT + 255) / 256, 256, 0, stream>>>(
        W_ix, W_ih, W_fx, W_fh, W_gx, W_gh, W_ox, W_oh, Wcat);
    k_packWcls<<<(HIDDEN * VOCAB + 255) / 256, 256, 0, stream>>>(W_c, Wcls);
    k_init    <<<(BATCH * HIDDEN + 255) / 256, 256, 0, stream>>>(cbuf, Apk);
    k_packAemb<<<(unsigned)(((size_t)SEQ * 64 * 512 + 255) / 256), 256, 0, stream>>>(
        x, embBf, Apk);

    for (int t = 0; t < SEQ; ++t) {
        k_step<<<64, 256, 0, stream>>>(Apk + (size_t)t * A_STRIDE,
                                       Apk + (size_t)(t + 1) * A_STRIDE,
                                       Wcat, bcat, cbuf);
    }

    k_cls<<<8192, 128, 0, stream>>>(Apk, Wcls, b_c, (float*)d_out);
}
